// AttnMethods_68985764708866
// MI455X (gfx1250) — compile-verified, matrix operand order checked by reading, not run on hardware
//
#include <hip/hip_runtime.h>
#include <hip/hip_bf16.h>

#define HEAD_NUM 32
#define HEAD_DIM 128
#define QS       128
#define KV_GPU   4096
#define KV_CPU   8192
#define BH       64          // BATCH * HEAD_NUM
#define PART     2048
#define NPART    7           // 2 gpu-cache + 1 new-kv(masked) + 4 cpu-cache
#define CHUNK    64          // keys per staged chunk
#define LOG2E_REF 1.4427f            // reference's constant for lse
#define LOG2E_EXACT 1.4426950408889634f

typedef __attribute__((ext_vector_type(16))) __bf16 v16bf;
typedef __attribute__((ext_vector_type(2)))  __bf16 v2bf;
typedef __attribute__((ext_vector_type(8)))  float  v8f;

struct U4x2 { uint4 lo, hi; };

static __device__ __forceinline__ unsigned pack2(float a, float b) {
#if __has_builtin(__builtin_amdgcn_cvt_pk_bf16_f32)
  auto r = __builtin_amdgcn_cvt_pk_bf16_f32(a, b);   // v_cvt_pk_bf16_f32
  return __builtin_bit_cast(unsigned, r);
#else
  // native bf16 casts; backend supports bf16 (WMMA operands), RNE rounding
  v2bf t;
  t[0] = (__bf16)a;
  t[1] = (__bf16)b;
  return __builtin_bit_cast(unsigned, t);
#endif
}
static __device__ __forceinline__ unsigned short f2bf_bits(float f) {
  __bf16 h = (__bf16)f;
  return __builtin_bit_cast(unsigned short, h);
}
static __device__ __forceinline__ v16bf make_ab(uint4 lo, uint4 hi) {
  U4x2 s{lo, hi};
  return __builtin_bit_cast(v16bf, s);
}
// raw v_exp_f32: args are always <= 0 here; large-negative flushes to 0, which
// is exactly the semantics softmax needs (no libcall range fixup required)
static __device__ __forceinline__ float fexp2(float x) {
#if __has_builtin(__builtin_amdgcn_exp2f)
  return __builtin_amdgcn_exp2f(x);
#else
  return exp2f(x);
#endif
}

// max-reduce across the 16-lane row this lane lives in (VALU permlane16)
static __device__ __forceinline__ float rowmax16(float x) {
#if __has_builtin(__builtin_amdgcn_permlane16)
  unsigned v, o;
  v = __builtin_bit_cast(unsigned, x);
  o = __builtin_amdgcn_permlane16(v, v, 0x67452301u, 0xEFCDAB89u, false, false); // xor 1
  x = fmaxf(x, __builtin_bit_cast(float, o)); v = __builtin_bit_cast(unsigned, x);
  o = __builtin_amdgcn_permlane16(v, v, 0x54761032u, 0xDCFE98BAu, false, false); // xor 2
  x = fmaxf(x, __builtin_bit_cast(float, o)); v = __builtin_bit_cast(unsigned, x);
  o = __builtin_amdgcn_permlane16(v, v, 0x32107654u, 0xBA98FEDCu, false, false); // xor 4
  x = fmaxf(x, __builtin_bit_cast(float, o)); v = __builtin_bit_cast(unsigned, x);
  o = __builtin_amdgcn_permlane16(v, v, 0xFEDCBA98u, 0x76543210u, false, false); // xor 8
  x = fmaxf(x, __builtin_bit_cast(float, o));
  return x;
#else
  x = fmaxf(x, __shfl_xor(x, 1));
  x = fmaxf(x, __shfl_xor(x, 2));
  x = fmaxf(x, __shfl_xor(x, 4));
  x = fmaxf(x, __shfl_xor(x, 8));
  return x;
#endif
}

// ---------------------------------------------------------------------------
// Kernel 1: partial flash attention over one KV partition.
// grid = (BH, NPART), block = 256 threads = 8 waves; wave w owns q-rows
// [16w, 16w+16). KV in 64-key chunks, register-double-buffered staging;
// per chunk: 16 WMMAs for S = Q*K^T, 16 for O += P*V, 2 for L += P*ones.
// ---------------------------------------------------------------------------
__global__ __launch_bounds__(256) void fa_partial_kernel(
    const float* __restrict__ q,  const float* __restrict__ knew,
    const float* __restrict__ vnew,
    const float* __restrict__ kg, const float* __restrict__ vg,
    const float* __restrict__ kc, const float* __restrict__ vc,
    float* __restrict__ outp, float* __restrict__ lsep)
{
  __shared__ __align__(16) unsigned short ldsK[CHUNK * HEAD_DIM];   // K chunk, row-major bf16
  __shared__ __align__(16) unsigned short ldsVT[HEAD_DIM * CHUNK];  // V chunk transposed bf16
  __shared__ __align__(16) unsigned short ldsP[8][16 * CHUNK];      // per-wave P scratch

  const int bh   = blockIdx.x;
  const int p    = blockIdx.y;
  const int tid  = threadIdx.x;
  const int wave = tid >> 5;
  const int lane = tid & 31;
  const int lm16 = lane & 15;
  const int half = lane >> 4;     // 0: K-chunks {0..7,16..23}; 1: {8..15,24..31}
  const int koff = half * 8;

  // ---- partition descriptor ----
  const float *ksrc, *vsrc;
  int kvbase, len, srcStride;
  bool masked = false;
  if (p < 2)        { ksrc = kg;   vsrc = vg;   kvbase = p * PART;       len = PART; srcStride = KV_GPU; }
  else if (p == 2)  { ksrc = knew; vsrc = vnew; kvbase = 0;              len = QS;   srcStride = QS; masked = true; }
  else              { ksrc = kc;   vsrc = vc;   kvbase = (p - 3) * PART; len = PART; srcStride = KV_CPU; }
  const float* kbase = ksrc + ((size_t)bh * srcStride + kvbase) * HEAD_DIM;
  const float* vbase = vsrc + ((size_t)bh * srcStride + kvbase) * HEAD_DIM;

  // ---- load Q tile into WMMA A-operand registers (16 rows x 128, bf16) ----
  const int qrow = wave * 16 + lm16;
  const float* qp = q + ((size_t)bh * QS + qrow) * HEAD_DIM;
  v16bf qa[4];
#pragma unroll
  for (int s = 0; s < 4; ++s) {
    const float4* c1 = (const float4*)(qp + s * 32 + koff);
    const float4* c2 = (const float4*)(qp + s * 32 + 16 + koff);
    float4 a0 = c1[0], a1 = c1[1], b0 = c2[0], b1 = c2[1];
    uint4 lo = make_uint4(pack2(a0.x, a0.y), pack2(a0.z, a0.w),
                          pack2(a1.x, a1.y), pack2(a1.z, a1.w));
    uint4 hi = make_uint4(pack2(b0.x, b0.y), pack2(b0.z, b0.w),
                          pack2(b1.x, b1.y), pack2(b1.z, b1.w));
    qa[s] = make_ab(lo, hi);
  }

  // all-ones bf16 B operand (row-sum trick): sum_k P[r][k] = (P*ones)[r][n]
  const uint4 ones4 = make_uint4(0x3F803F80u, 0x3F803F80u, 0x3F803F80u, 0x3F803F80u);
  const v16bf onesv = make_ab(ones4, ones4);

  // ---- online-softmax state ----
  v8f O[8];
#pragma unroll
  for (int t = 0; t < 8; ++t) O[t] = (v8f){0,0,0,0,0,0,0,0};
  v8f L = (v8f){0,0,0,0,0,0,0,0};            // row-sum accumulator
  float m_i[8];
#pragma unroll
  for (int r = 0; r < 8; ++r) m_i[r] = -3.0e38f;

  const int nch = len / CHUNK;

  // ---- register double buffer for the staged K/V chunk ----
  float4 kreg[8], vreg[8];
  {
    const float4* k4 = (const float4*)kbase;
    const float4* v4 = (const float4*)vbase;
#pragma unroll
    for (int i = 0; i < 8; ++i) { kreg[i] = k4[tid + i * 256]; vreg[i] = v4[tid + i * 256]; }
  }

  for (int ch = 0; ch < nch; ++ch) {
    __syncthreads();
    // ---- convert+store staged regs: K -> ldsK, V -> ldsVT (transposed) ----
#pragma unroll
    for (int i = 0; i < 8; ++i) {
      int e = tid + i * 256;                  // 2048 float4 per 64x128 tile
      float4 kk = kreg[i];
      *(uint2*)&ldsK[e * 4] = make_uint2(pack2(kk.x, kk.y), pack2(kk.z, kk.w));
      float4 vv = vreg[i];
      int le = e * 4, kr = le >> 7, nn = le & 127;
      unsigned vxy = pack2(vv.x, vv.y), vzw = pack2(vv.z, vv.w);
      ldsVT[(nn + 0) * CHUNK + kr] = (unsigned short)(vxy & 0xffffu);
      ldsVT[(nn + 1) * CHUNK + kr] = (unsigned short)(vxy >> 16);
      ldsVT[(nn + 2) * CHUNK + kr] = (unsigned short)(vzw & 0xffffu);
      ldsVT[(nn + 3) * CHUNK + kr] = (unsigned short)(vzw >> 16);
    }
    __syncthreads();

    // ---- issue next chunk's global loads now; they complete during compute ----
    if (ch + 1 < nch) {
      const float4* k4 = (const float4*)(kbase + (size_t)(ch + 1) * CHUNK * HEAD_DIM);
      const float4* v4 = (const float4*)(vbase + (size_t)(ch + 1) * CHUNK * HEAD_DIM);
#pragma unroll
      for (int i = 0; i < 8; ++i) { kreg[i] = k4[tid + i * 256]; vreg[i] = v4[tid + i * 256]; }
    }
    if (ch + 2 < nch) {                       // gfx1250 global_prefetch_b8 -> L2
      __builtin_prefetch(kbase + (size_t)(ch + 2) * CHUNK * HEAD_DIM + tid * 32, 0, 1);
      __builtin_prefetch(vbase + (size_t)(ch + 2) * CHUNK * HEAD_DIM + tid * 32, 0, 1);
    }

    // ---- S = Q * K^T : four 16x16 key tiles, 4 chained K=32 WMMAs each ----
    v8f sacc[4];
#pragma unroll
    for (int t = 0; t < 4; ++t) {
      v8f c = (v8f){0,0,0,0,0,0,0,0};
      const int krow = t * 16 + lm16;         // B lane n = key row
#pragma unroll
      for (int s = 0; s < 4; ++s) {
        uint4 b1 = *(const uint4*)&ldsK[krow * HEAD_DIM + s * 32 + koff];
        uint4 b2 = *(const uint4*)&ldsK[krow * HEAD_DIM + s * 32 + 16 + koff];
        v16bf kb = make_ab(b1, b2);
        c = __builtin_amdgcn_wmma_f32_16x16x32_bf16(false, qa[s], false, kb,
                                                    (short)0, c, false, false);
      }
      sacc[t] = c;
    }

    // ---- analytic causal mask (new-kv partition only) ----
    if (masked) {
#pragma unroll
      for (int t = 0; t < 4; ++t)
#pragma unroll
        for (int r = 0; r < 8; ++r) {
          int col  = ch * CHUNK + t * 16 + lm16;
          int rowg = wave * 16 + r + half * 8;
          if (col > rowg) sacc[t][r] += -65504.0f;
        }
    }

    // ---- online softmax: permlane16 max; rescale only when max moved ----
    float nm[8];
    bool upd = false;
#pragma unroll
    for (int r = 0; r < 8; ++r) {
      float cm = fmaxf(fmaxf(sacc[0][r], sacc[1][r]),
                       fmaxf(sacc[2][r], sacc[3][r]));
      cm = rowmax16(cm);
      nm[r] = fmaxf(m_i[r], cm);
      upd = upd || (nm[r] > m_i[r]);
    }
    if (__any(upd)) {                         // wave-uniform skip of rescale
#pragma unroll
      for (int r = 0; r < 8; ++r) {
        float sc = fexp2((m_i[r] - nm[r]) * LOG2E_EXACT);
#pragma unroll
        for (int t = 0; t < 8; ++t) O[t][r] *= sc;
        L[r] *= sc;
      }
    }
#pragma unroll
    for (int r = 0; r < 8; ++r) {
      m_i[r] = nm[r];
#pragma unroll
      for (int t = 0; t < 4; ++t)
        sacc[t][r] = fexp2((sacc[t][r] - nm[r]) * LOG2E_EXACT);
    }

    // ---- transpose P (C-layout -> A-layout) through per-wave LDS scratch ----
    unsigned short* pw = ldsP[wave];
#pragma unroll
    for (int t = 0; t < 4; ++t)
#pragma unroll
      for (int r = 0; r < 8; ++r)
        pw[(r + half * 8) * CHUNK + t * 16 + lm16] = f2bf_bits(sacc[t][r]);
    asm volatile("s_wait_dscnt 0" ::: "memory");   // CDNA5 split DS counter
    v16bf pa[2];
#pragma unroll
    for (int g = 0; g < 2; ++g) {
      uint4 p1v = *(const uint4*)&pw[lm16 * CHUNK + g * 32 + koff];
      uint4 p2v = *(const uint4*)&pw[lm16 * CHUNK + g * 32 + 16 + koff];
      pa[g] = make_ab(p1v, p2v);
    }

    // ---- O += P * V (8 d-tiles x 2 key-halves), L += P * ones ----
#pragma unroll
    for (int g = 0; g < 2; ++g) {
#pragma unroll
      for (int t = 0; t < 8; ++t) {
        const int n = t * 16 + lm16;
        uint4 vb1 = *(const uint4*)&ldsVT[n * CHUNK + g * 32 + koff];
        uint4 vb2 = *(const uint4*)&ldsVT[n * CHUNK + g * 32 + 16 + koff];
        v16bf vb = make_ab(vb1, vb2);
        O[t] = __builtin_amdgcn_wmma_f32_16x16x32_bf16(false, pa[g], false, vb,
                                                       (short)0, O[t], false, false);
      }
      L = __builtin_amdgcn_wmma_f32_16x16x32_bf16(false, pa[g], false, onesv,
                                                  (short)0, L, false, false);
    }
  }

  // ---- finalize partial state: normalized O and base-2 LSE ----
  float* op = outp + ((size_t)p * BH + bh) * QS * HEAD_DIM;
  float inv[8];
#pragma unroll
  for (int r = 0; r < 8; ++r) inv[r] = 1.0f / L[r];
#pragma unroll
  for (int t = 0; t < 8; ++t)
#pragma unroll
    for (int r = 0; r < 8; ++r) {
      int rowg = wave * 16 + r + half * 8;
      op[(size_t)rowg * HEAD_DIM + t * 16 + lm16] = O[t][r] * inv[r];
    }
  if (lm16 == 0) {
#pragma unroll
    for (int r = 0; r < 8; ++r) {
      int rowg = wave * 16 + r + half * 8;
      lsep[((size_t)p * BH + bh) * QS + rowg] = (logf(L[r]) + m_i[r]) * LOG2E_REF;
    }
  }
}

// ---------------------------------------------------------------------------
// Kernel 2: merge partial states (flashinfer merge_state, lse in log2) and
// permute [bh,q,d] -> [b,q,h,d].
// ---------------------------------------------------------------------------
__global__ __launch_bounds__(256) void merge_kernel(
    const float* __restrict__ outp, const float* __restrict__ lsep,
    float* __restrict__ out)
{
  int idx = blockIdx.x * 256 + threadIdx.x;     // over BH*QS*HEAD_DIM
  if (idx >= BH * QS * HEAD_DIM) return;
  int d  = idx & 127;
  int qq = (idx >> 7) & 127;
  int bh = idx >> 14;

  float lse[NPART], mmax = -3.0e38f;
#pragma unroll
  for (int p = 0; p < NPART; ++p) {
    lse[p] = lsep[((size_t)p * BH + bh) * QS + qq];
    mmax = fmaxf(mmax, lse[p]);
  }
  float acc = 0.0f, wsum = 0.0f;
#pragma unroll
  for (int p = 0; p < NPART; ++p) {
    float w = fexp2(lse[p] - mmax);
    acc += w * outp[(((size_t)p * BH + bh) * QS + qq) * HEAD_DIM + d];
    wsum += w;
  }
  int b = bh >> 5, h = bh & 31;
  out[(((size_t)b * QS + qq) * HEAD_NUM + h) * HEAD_DIM + d] = acc / wsum;
}

extern "C" void kernel_launch(void* const* d_in, const int* in_sizes, int n_in,
                              void* d_out, int out_size, void* d_ws, size_t ws_size,
                              hipStream_t stream) {
  const float* q    = (const float*)d_in[0];
  const float* k    = (const float*)d_in[1];
  const float* v    = (const float*)d_in[2];
  const float* kg   = (const float*)d_in[3];
  const float* vg   = (const float*)d_in[4];
  const float* kc   = (const float*)d_in[5];
  const float* vc   = (const float*)d_in[6];
  // d_in[7] = additive causal mask; applied analytically in-kernel.

  float* ws   = (float*)d_ws;
  float* outp = ws;                                        // NPART*BH*QS*D floats
  float* lsep = ws + (size_t)NPART * BH * QS * HEAD_DIM;   // NPART*BH*QS floats

  dim3 g1(BH, NPART);
  fa_partial_kernel<<<g1, 256, 0, stream>>>(q, k, v, kg, vg, kc, vc, outp, lsep);

  int total = BH * QS * HEAD_DIM;
  merge_kernel<<<(total + 255) / 256, 256, 0, stream>>>(outp, lsep, (float*)d_out);
}
